// SERESkippedQwen3MoeSparseMoeBlock_87935160418449
// MI455X (gfx1250) — matrix operand order repaired and checked by reading, not compiled, not run on hardware
//
#include <hip/hip_runtime.h>

// ---------------- problem constants (from reference) ----------------
#define N_TOK 2048      // B*SEQ
#define HDIM  1024
#define FDIM  1024
#define NEXP  8
#define TOPK  2
#define THRESH 0.7f

#define KSTEP 64        // K elements staged per double-buffer step
#define PITCH 72        // padded LDS row pitch in elements (144B, 16B-mult)

// ---------------- raw vector types (avoid HIP struct vector ctors) --
typedef __attribute__((ext_vector_type(16))) __bf16         v16bf;
typedef __attribute__((ext_vector_type(8)))  float          v8f;
typedef __attribute__((ext_vector_type(4)))  unsigned int   u32x4;
typedef __attribute__((ext_vector_type(4)))  float          f32x4;
typedef __attribute__((ext_vector_type(4)))  unsigned short u16x4;

union FragCast { v16bf v; u32x4 q[2]; };

// fp32 -> bf16 round-to-nearest-even
__device__ __forceinline__ unsigned short f2bf(float f) {
    unsigned int u = __float_as_uint(f);
    u += 0x7FFFu + ((u >> 16) & 1u);
    return (unsigned short)(u >> 16);
}

// ---------------- CDNA5 async copy global->LDS (ASYNCcnt) -----------
// global_load_async_to_lds_b128: vdst = 32-bit LDS byte address VGPR,
// vaddr = 64-bit global address. Tracked with ASYNCcnt (ISA ch.08).
__device__ __forceinline__ void async_load_b128(unsigned lds_off, const void* src) {
    asm volatile("global_load_async_to_lds_b128 %0, %1, off"
                 :: "v"(lds_off), "v"(src)
                 : "memory");
}

template <int N>
__device__ __forceinline__ void wait_async() {
    asm volatile("s_wait_asynccnt %0" :: "n"(N) : "memory");
}

// generic pointer to LDS: aperture lives in bits [63:32], so the low
// 32 bits are the LDS byte offset the hardware wants in VDST.
__device__ __forceinline__ unsigned lds_addr(const void* p) {
    return (unsigned)(size_t)p;
}

// ---------------- WMMA fragment loaders (CDNA5 wave32 layouts) ------
// A matrix 16x32 bf16 (MxK): lanes 0-15: M=lane, K={k0..k0+7, k0+16..k0+23}
//                            lanes 16-31: M=lane-16, K={k0+8..k0+15, k0+24..k0+31}
__device__ __forceinline__ v16bf load_a(const unsigned short* M, int ld,
                                        int row0, int k0, int lane) {
    int r  = row0 + (lane & 15);
    int kb = k0 + ((lane >> 4) << 3);       // +8 for hi half-wave
    const unsigned short* p = M + (size_t)r * ld + kb;
    FragCast u;
    u.q[0] = *(const u32x4*)(p);            // K kb .. kb+7
    u.q[1] = *(const u32x4*)(p + 16);       // K kb+16 .. kb+23
    return u.v;
}

// B matrix 32x16 bf16 (KxN), B(k,n) = W[col0+n, k] with W row-major [*, ld]:
// lanes 0-15: N=lane, K={k0..k0+7, k0+16..k0+23} ; lanes 16-31: N=lane-16, K={k0+8..k0+15, k0+24..k0+31}
__device__ __forceinline__ v16bf load_b(const unsigned short* W, int ld,
                                        int col0, int k0, int lane) {
    int c  = col0 + (lane & 15);
    int kb = k0 + ((lane >> 4) << 3);       // +8 for hi half-wave
    const unsigned short* p = W + (size_t)c * ld + kb;
    FragCast u;
    u.q[0] = *(const u32x4*)(p);
    u.q[1] = *(const u32x4*)(p + 16);
    return u.v;
}

__device__ __forceinline__ v8f wmma_bf16(v16bf a, v16bf b, v8f c) {
    // (neg_a, A, neg_b, B, c_mod, C, reuse_a, reuse_b)
    return __builtin_amdgcn_wmma_f32_16x16x32_bf16(
        false, a, false, b, (short)0, c, false, false);
}

// ---------------- small kernels -------------------------------------
__global__ void init_kernel(unsigned* mask) {
    if (threadIdx.x == 0) *mask = 0u;
}

__global__ void cvt_kernel(const float* __restrict__ src,
                           unsigned short* __restrict__ dst, int n4) {
    int i = blockIdx.x * blockDim.x + threadIdx.x;
    if (i >= n4) return;
    f32x4 f = ((const f32x4*)src)[i];
    u16x4 o;
    o.x = f2bf(f.x); o.y = f2bf(f.y); o.z = f2bf(f.z); o.w = f2bf(f.w);
    ((u16x4*)dst)[i] = o;
}

// one wave (32 lanes) per token: logits -> softmax(fp32) -> top2 (+norm)
__global__ __launch_bounds__(256)
void router_kernel(const float* __restrict__ x, const float* __restrict__ gw,
                   float* __restrict__ topw, int* __restrict__ topi,
                   unsigned* __restrict__ mask) {
    const int wave = threadIdx.x >> 5;
    const int lane = threadIdx.x & 31;
    const int n = blockIdx.x * 8 + wave;

    float acc[NEXP];
#pragma unroll
    for (int e = 0; e < NEXP; ++e) acc[e] = 0.f;

    for (int h = lane; h < HDIM; h += 32) {
        float xv = x[(size_t)n * HDIM + h];
#pragma unroll
        for (int e = 0; e < NEXP; ++e)
            acc[e] += xv * gw[(size_t)e * HDIM + h];
    }
#pragma unroll
    for (int off = 16; off > 0; off >>= 1) {
#pragma unroll
        for (int e = 0; e < NEXP; ++e)
            acc[e] += __shfl_xor(acc[e], off, 32);
    }
    if (lane == 0) {
        float mx = acc[0];
#pragma unroll
        for (int e = 1; e < NEXP; ++e) mx = fmaxf(mx, acc[e]);
        float p[NEXP], s = 0.f;
#pragma unroll
        for (int e = 0; e < NEXP; ++e) { p[e] = __expf(acc[e] - mx); s += p[e]; }
#pragma unroll
        for (int e = 0; e < NEXP; ++e) p[e] /= s;

        int i0 = 0;
#pragma unroll
        for (int e = 1; e < NEXP; ++e) if (p[e] > p[i0]) i0 = e;
        int i1 = (i0 == 0) ? 1 : 0;
#pragma unroll
        for (int e = 0; e < NEXP; ++e)
            if (e != i0 && p[e] > p[i1]) i1 = e;

        float w0 = p[i0], w1 = p[i1];
        float sum = fmaxf(w0 + w1, 1e-12f);
        topw[n * 2 + 0] = w0 / sum;
        topw[n * 2 + 1] = w1 / sum;
        topi[n * 2 + 0] = i0;
        topi[n * 2 + 1] = i1;
        atomicOr(mask, 1u << i0);   // primary = union of top-1 experts
    }
}

// expert_map[e]: non-primary experts redirect to most-similar primary
__global__ void reroute_kernel(const unsigned* __restrict__ mask,
                               const float* __restrict__ sim,
                               int* __restrict__ emap) {
    int e = threadIdx.x;
    if (e >= NEXP) return;
    unsigned pm = *mask;
    float best = -1e30f; int bj = 0; bool found = false;
#pragma unroll
    for (int j = 0; j < NEXP; ++j) {
        if ((pm >> j) & 1u) {
            float s = sim[e * NEXP + j];
            if (!found || s > best) { best = s; bj = j; found = true; }
        }
    }
    bool isPrim = (pm >> e) & 1u;
    emap[e] = (!isPrim && found && best >= THRESH) ? bj : e;
}

// dense combine table [N, E] (scatter-add of rerouted top-2 weights)
__global__ void combine_kernel(const float* __restrict__ topw,
                               const int* __restrict__ topi,
                               const int* __restrict__ emap,
                               float* __restrict__ comb) {
    int n = blockIdx.x * blockDim.x + threadIdx.x;
    if (n >= N_TOK) return;
    float c[NEXP];
#pragma unroll
    for (int e = 0; e < NEXP; ++e) c[e] = 0.f;
#pragma unroll
    for (int k = 0; k < TOPK; ++k) {
        int ei = emap[topi[n * 2 + k]];
        float w = topw[n * 2 + k];
#pragma unroll
        for (int e = 0; e < NEXP; ++e)
            c[e] += (ei == e) ? w : 0.f;
    }
#pragma unroll
    for (int e = 0; e < NEXP; ++e) comb[(size_t)n * NEXP + e] = c[e];
}

// ---------------- GEMM 1: hmid[e] = silu(x Wg^T) * (x Wu^T) ---------
// block = 256 thr = 8 waves -> 128(M) x 64(F-cols) tile. Double-buffered
// async global->LDS staging (ASYNCcnt); wave computes 32x32 g + 32x32 u
// (16 WMMAs per staged K=64 step). Issue path is branch-free: the A/B
// split is on the unrolled compile-time index g, so each wave issues
// exactly 8 async ops per step and s_wait_asynccnt 8 is exact.
__global__ __launch_bounds__(256)
void gemm1_kernel(const unsigned short* __restrict__ xb,     // [N,H] bf16
                  const unsigned short* __restrict__ gupb,   // [E,2F,H] bf16
                  unsigned short* __restrict__ hmid) {       // [E,N,F] bf16
    __shared__ __align__(16) unsigned short shA[2][128 * PITCH]; // 2 x 18KB
    __shared__ __align__(16) unsigned short shB[2][128 * PITCH]; // 2 x 18KB

    const int e    = blockIdx.z;
    const int wave = threadIdx.x >> 5;
    const int lane = threadIdx.x & 31;
    const int mblk = blockIdx.y * 128;
    const int jblk = blockIdx.x * 64;
    const unsigned short* W = gupb + (size_t)e * (2 * FDIM) * HDIM;

    // staging plan: per step 64 async b128 wave-ops (32 A + 32 B), 8 per
    // wave (g<4 -> A, g>=4 -> B); each op moves 4 rows x 128B of a tile.
    const int sub = lane >> 3;          // row within op (0..3)
    const int boe = (lane & 7) * 8;     // element offset in row (16B chunks)

    auto issue = [&](int buf, int k0) {
#pragma unroll
        for (int g = 0; g < 8; ++g) {
            if (g < 4) {                              // compile-time split
                int row = (wave * 4 + g) * 4 + sub;   // 0..127
                const unsigned short* src =
                    xb + (size_t)(mblk + row) * HDIM + k0 + boe;
                async_load_b128(lds_addr(&shA[buf][row * PITCH + boe]), src);
            } else {
                int row = (wave * 4 + (g - 4)) * 4 + sub;   // 0..127
                // rows 0..63 = gate cols jblk+row ; 64..127 = up cols
                int grow = jblk + row + ((row >> 6) * (FDIM - 64));
                const unsigned short* src =
                    W + (size_t)grow * HDIM + k0 + boe;
                async_load_b128(lds_addr(&shB[buf][row * PITCH + boe]), src);
            }
        }
    };

    v8f accg[2][2] = {};
    v8f accu[2][2] = {};
    const int ml = (wave & 3) * 32;
    const int jl = (wave >> 2) * 32;

    issue(0, 0);
    const int nsteps = HDIM / KSTEP;   // 16
    for (int it = 0; it < nsteps; ++it) {
        int cur = it & 1;
        if (it + 1 < nsteps) { issue(cur ^ 1, (it + 1) * KSTEP); wait_async<8>(); }
        else                 { wait_async<0>(); }
        __syncthreads();
        const unsigned short* As = &shA[cur][0];
        const unsigned short* Bs = &shB[cur][0];
#pragma unroll
        for (int ks = 0; ks < KSTEP; ks += 32) {
            v16bf a0  = load_a(As, PITCH, ml,      ks, lane);
            v16bf a1  = load_a(As, PITCH, ml + 16, ks, lane);
            v16bf bg0 = load_b(Bs, PITCH, jl,           ks, lane);
            v16bf bg1 = load_b(Bs, PITCH, jl + 16,      ks, lane);
            v16bf bu0 = load_b(Bs, PITCH, 64 + jl,      ks, lane);
            v16bf bu1 = load_b(Bs, PITCH, 64 + jl + 16, ks, lane);
            accg[0][0] = wmma_bf16(a0, bg0, accg[0][0]);
            accg[0][1] = wmma_bf16(a0, bg1, accg[0][1]);
            accg[1][0] = wmma_bf16(a1, bg0, accg[1][0]);
            accg[1][1] = wmma_bf16(a1, bg1, accg[1][1]);
            accu[0][0] = wmma_bf16(a0, bu0, accu[0][0]);
            accu[0][1] = wmma_bf16(a0, bu1, accu[0][1]);
            accu[1][0] = wmma_bf16(a1, bu0, accu[1][0]);
            accu[1][1] = wmma_bf16(a1, bu1, accu[1][1]);
        }
        __syncthreads();   // protect buffer reuse by the next-next issue
    }

    unsigned short* dst = hmid + (size_t)e * N_TOK * FDIM;
#pragma unroll
    for (int mi = 0; mi < 2; ++mi)
#pragma unroll
        for (int ni = 0; ni < 2; ++ni)
#pragma unroll
            for (int v = 0; v < 8; ++v) {
                float g = accg[mi][ni][v];
                float u = accu[mi][ni][v];
                float h = (g / (1.f + __expf(-g))) * u;   // silu(g)*u
                int m = mblk + ml + mi * 16 + v + ((lane >> 4) << 3);
                int n = jblk + jl + ni * 16 + (lane & 15);
                dst[(size_t)m * FDIM + n] = f2bf(h);
            }
}

// ---------------- GEMM 2: out = sum_e comb[:,e] * (hmid[e] Wd[e]^T) --
// expert loop inside block -> single fp32 store, no atomics. Same
// double-buffered async staging; 48 async ops/step, exactly 6 per wave
// (g<4 -> A, g>=4 -> B; compile-time split).
__global__ __launch_bounds__(256)
void gemm2_kernel(const unsigned short* __restrict__ hmid,  // [E,N,F] bf16
                  const unsigned short* __restrict__ dwnb,  // [E,H,F] bf16
                  const float* __restrict__ comb,           // [N,E]
                  float* __restrict__ out) {                // [N,H]
    __shared__ __align__(16) unsigned short shA[2][128 * PITCH]; // 2 x 18KB
    __shared__ __align__(16) unsigned short shB[2][64 * PITCH];  // 2 x 9KB

    const int wave = threadIdx.x >> 5;
    const int lane = threadIdx.x & 31;
    const int mblk = blockIdx.y * 128;
    const int hblk = blockIdx.x * 64;
    const int sub = lane >> 3;
    const int boe = (lane & 7) * 8;

    auto issue = [&](int buf, int s) {
        int e  = s >> 4;                // FDIM/KSTEP = 16 steps per expert
        int k0 = (s & 15) * KSTEP;
        const unsigned short* A  = hmid + (size_t)e * N_TOK * FDIM;
        const unsigned short* Wd = dwnb + (size_t)e * HDIM * FDIM;
#pragma unroll
        for (int g = 0; g < 6; ++g) {
            if (g < 4) {                              // compile-time split
                int row = (wave * 4 + g) * 4 + sub;   // 0..127
                const unsigned short* src =
                    A + (size_t)(mblk + row) * FDIM + k0 + boe;
                async_load_b128(lds_addr(&shA[buf][row * PITCH + boe]), src);
            } else {
                int row = (wave * 2 + (g - 4)) * 4 + sub;   // 0..63
                const unsigned short* src =
                    Wd + (size_t)(hblk + row) * FDIM + k0 + boe;
                async_load_b128(lds_addr(&shB[buf][row * PITCH + boe]), src);
            }
        }
    };

    const int ml = (wave & 3) * 32;
    const int hl = (wave >> 2) * 32;

    v8f oacc[2][2] = {};
    v8f acc[2][2] = {};

    issue(0, 0);
    const int tot = NEXP * (FDIM / KSTEP);   // 128 steps
    for (int s = 0; s < tot; ++s) {
        int cur = s & 1;
        if (s + 1 < tot) { issue(cur ^ 1, s + 1); wait_async<6>(); }
        else             { wait_async<0>(); }
        __syncthreads();
        const unsigned short* As = &shA[cur][0];
        const unsigned short* Bs = &shB[cur][0];
#pragma unroll
        for (int ks = 0; ks < KSTEP; ks += 32) {
            v16bf a0 = load_a(As, PITCH, ml,      ks, lane);
            v16bf a1 = load_a(As, PITCH, ml + 16, ks, lane);
            v16bf b0 = load_b(Bs, PITCH, hl,      ks, lane);
            v16bf b1 = load_b(Bs, PITCH, hl + 16, ks, lane);
            acc[0][0] = wmma_bf16(a0, b0, acc[0][0]);
            acc[0][1] = wmma_bf16(a0, b1, acc[0][1]);
            acc[1][0] = wmma_bf16(a1, b0, acc[1][0]);
            acc[1][1] = wmma_bf16(a1, b1, acc[1][1]);
        }
        __syncthreads();

        if ((s & 15) == 15) {           // end of this expert's K loop
            int e = s >> 4;
#pragma unroll
            for (int mi = 0; mi < 2; ++mi)
#pragma unroll
                for (int ni = 0; ni < 2; ++ni) {
#pragma unroll
                    for (int v = 0; v < 8; ++v) {
                        int m = mblk + ml + mi * 16 + v + ((lane >> 4) << 3);
                        float w = comb[(size_t)m * NEXP + e];
                        oacc[mi][ni][v] += w * acc[mi][ni][v];
                    }
                    v8f z = {};
                    acc[mi][ni] = z;
                }
        }
    }

#pragma unroll
    for (int mi = 0; mi < 2; ++mi)
#pragma unroll
        for (int ni = 0; ni < 2; ++ni)
#pragma unroll
            for (int v = 0; v < 8; ++v) {
                int m = mblk + ml + mi * 16 + v + ((lane >> 4) << 3);
                int h = hblk + hl + ni * 16 + (lane & 15);
                out[(size_t)m * HDIM + h] = oacc[mi][ni][v];
            }
}

// ---------------- launch --------------------------------------------
extern "C" void kernel_launch(void* const* d_in, const int* in_sizes, int n_in,
                              void* d_out, int out_size, void* d_ws, size_t ws_size,
                              hipStream_t stream) {
    (void)in_sizes; (void)n_in; (void)out_size; (void)ws_size;
    const float* x   = (const float*)d_in[0];   // [2,1024,1024]
    const float* gw  = (const float*)d_in[1];   // [8,1024]
    const float* gup = (const float*)d_in[2];   // [8,2048,1024]
    const float* dwn = (const float*)d_in[3];   // [8,1024,1024]
    const float* sim = (const float*)d_in[4];   // [8,8]
    float* out = (float*)d_out;

    char* ws = (char*)d_ws;
    size_t off = 0;
    auto alloc = [&](size_t bytes) -> void* {
        void* p = (void*)(ws + off);
        off += (bytes + 255) & ~(size_t)255;
        return p;
    };
    unsigned short* xb   = (unsigned short*)alloc((size_t)N_TOK * HDIM * 2);          //  4 MB
    unsigned short* gupb = (unsigned short*)alloc((size_t)NEXP * 2 * FDIM * HDIM * 2);// 32 MB
    unsigned short* dwnb = (unsigned short*)alloc((size_t)NEXP * HDIM * FDIM * 2);    // 16 MB
    unsigned short* hmid = (unsigned short*)alloc((size_t)NEXP * N_TOK * FDIM * 2);   // 32 MB
    float*    topw = (float*)alloc((size_t)N_TOK * TOPK * 4);
    int*      topi = (int*)alloc((size_t)N_TOK * TOPK * 4);
    float*    comb = (float*)alloc((size_t)N_TOK * NEXP * 4);
    unsigned* mask = (unsigned*)alloc(4);
    int*      emap = (int*)alloc(NEXP * 4);

    init_kernel<<<1, 32, 0, stream>>>(mask);

    { int n4 = N_TOK * HDIM / 4;
      cvt_kernel<<<(n4 + 255) / 256, 256, 0, stream>>>(x, xb, n4); }
    { int n4 = NEXP * 2 * FDIM * HDIM / 4;
      cvt_kernel<<<(n4 + 255) / 256, 256, 0, stream>>>(gup, gupb, n4); }
    { int n4 = NEXP * HDIM * FDIM / 4;
      cvt_kernel<<<(n4 + 255) / 256, 256, 0, stream>>>(dwn, dwnb, n4); }

    router_kernel<<<N_TOK / 8, 256, 0, stream>>>(x, gw, topw, topi, mask);
    reroute_kernel<<<1, 32, 0, stream>>>(mask, sim, emap);
    combine_kernel<<<N_TOK / 256, 256, 0, stream>>>(topw, topi, emap, comb);

    dim3 g1(FDIM / 64, N_TOK / 128, NEXP);
    gemm1_kernel<<<g1, 256, 0, stream>>>(xb, gupb, hmid);

    dim3 g2(HDIM / 64, N_TOK / 128);
    gemm2_kernel<<<g2, 256, 0, stream>>>(hmid, dwnb, comb, out);
}
